// TreeLSTM_45054206935075
// MI455X (gfx1250) — matrix operand order repaired and checked by reading, not compiled, not run on hardware
//
#include <hip/hip_runtime.h>
#include <hip/hip_bf16.h>

typedef __attribute__((ext_vector_type(16))) _Float16 v16h;
typedef __attribute__((ext_vector_type(8)))  float    v8f;
typedef _Float16 half_t;

#define L_LEAF 131072
#define EMB    300
#define HDIM   128
#define T_STEPS 50
#define KLEAF  320          // EMB padded to multiple of 32
#define NLEAF  384          // [i|o|u] outputs
#define KTREE  256          // [lh|rh]  (tt folded into bias)
#define NTREE  640          // [i|o|u|f1|f2]
#define AP_STRIDE 328       // KLEAF + 8 halfs (even -> dword aligned rows)
#define TOT_ROWS 262143     // 131072 + 65536 + ... + 1

__device__ __forceinline__ float sigf(float x) { return 1.0f / (1.0f + __expf(-x)); }

// ---- WMMA fragment loaders (CDNA5 16x16x32 f16 layouts, wave32) -----------

// A: 16x32 f16. lanes 0-15: M=lane, K pairs {0..7,16..23}; lanes 16-31: same M, K {8..15,24..31}
__device__ __forceinline__ v16h load_afrag_lds(const half_t* Ap, int m0, int K0, int lane) {
  int l = lane & 15, hi = lane >> 4;
  const half_t* rowp = Ap + (m0 + l) * AP_STRIDE;
  union { v16h v; unsigned u[8]; } r;
#pragma unroll
  for (int j = 0; j < 8; ++j) {
    int kOff = ((j < 4) ? 0 : 16) + hi * 8 + ((j & 3) << 1);
    r.u[j] = *(const unsigned*)(rowp + K0 + kOff);
  }
  return r.v;
}

__device__ __forceinline__ v16h load_afrag_global(const half_t* base, int row_stride,
                                                  int m0, int K0, int lane, int nvalid) {
  int l = lane & 15, hi = lane >> 4;
  int row = m0 + l;
  union { v16h v; unsigned u[8]; } r;
  if (row < nvalid) {
    const half_t* rowp = base + (size_t)row * row_stride;
#pragma unroll
    for (int j = 0; j < 8; ++j) {
      int kOff = ((j < 4) ? 0 : 16) + hi * 8 + ((j & 3) << 1);
      r.u[j] = *(const unsigned*)(rowp + K0 + kOff);
    }
  } else {
#pragma unroll
    for (int j = 0; j < 8; ++j) r.u[j] = 0u;
  }
  return r.v;
}

// B: 32x16 f16 row-major [K][N]. N=n0+lane%16; elements K {0..7,16..23} (lanes 0-15), {8..15,24..31} (lanes 16-31)
__device__ __forceinline__ v16h load_bfrag_global(const half_t* B, int ldb,
                                                  int K0, int n0, int lane) {
  int l = lane & 15, hi = lane >> 4;
  const half_t* p = B + n0 + l;
  v16h r;
#pragma unroll
  for (int s = 0; s < 16; ++s) {
    int kOff = ((s < 8) ? 0 : 8) + hi * 8 + s;
    r[s] = p[(size_t)(K0 + kOff) * ldb];
  }
  return r;
}

__device__ __forceinline__ v8f wmma_f16(v16h a, v16h b, v8f c) {
  return __builtin_amdgcn_wmma_f32_16x16x32_f16(false, a, false, b, (short)0, c, false, false);
}

// ---- embedding table -> f16 (once; makes leaf staging pure packed copies) --

__global__ void emb_to_f16_kernel(const float* emb, half_t* embh, int total2) {
  int idx = blockIdx.x * blockDim.x + threadIdx.x;
  if (idx >= total2) return;
  half_t h0 = (half_t)emb[2 * (size_t)idx];
  half_t h1 = (half_t)emb[2 * (size_t)idx + 1];
  union { half_t h[2]; unsigned u; } p;
  p.h[0] = h0; p.h[1] = h1;
  ((unsigned*)embh)[idx] = p.u;
}

// ---- weight packing --------------------------------------------------------

__global__ void pack_leaf_kernel(const float* Wi, const float* Wo, const float* Wu,
                                 const float* bi, const float* bo, const float* bu,
                                 half_t* Wpack, float* bias_leaf) {
  int idx = blockIdx.x * blockDim.x + threadIdx.x;
  if (idx >= KLEAF * NLEAF) return;
  int k = idx / NLEAF, n = idx - k * NLEAF;
  int g = n >> 7, c = n & 127;
  const float* W = (g == 0) ? Wi : ((g == 1) ? Wo : Wu);
  float v = (k < EMB) ? W[c * EMB + k] : 0.0f;
  Wpack[(size_t)k * NLEAF + n] = (half_t)v;
  if (k == 0) {
    const float* B = (g == 0) ? bi : ((g == 1) ? bo : bu);
    bias_leaf[n] = B[c];
  }
}

__global__ void pack_tree_kernel(const float* Ui, const float* Uo, const float* Uu,
                                 const float* Uf1, const float* Uf2, half_t* Upack) {
  int idx = blockIdx.x * blockDim.x + threadIdx.x;
  if (idx >= KTREE * NTREE) return;
  int k = idx / NTREE, n = idx - k * NTREE;
  float v = 0.0f;
  if (n < 384) {                       // i|o|u blocks use [lh|rh] = K 0..255
    int g = n >> 7, c = n & 127;
    const float* W = (g == 0) ? Ui : ((g == 1) ? Uo : Uu);
    v = W[c * 384 + k];
  } else if (n < 512) {                // f1 uses lh only (K<128)
    if (k < HDIM) v = Uf1[(n - 384) * HDIM + k];
  } else {                             // f2 uses rh only (K>=128)
    if (k >= HDIM) v = Uf2[(n - 512) * HDIM + (k - HDIM)];
  }
  Upack[(size_t)k * NTREE + n] = (half_t)v;
}

// ---- small sequential LSTM for th ------------------------------------------

__global__ __launch_bounds__(512) void lstm_kernel(const float* emb, const int* targets,
                                                   const float* wih, const float* whh,
                                                   const float* bih, const float* bhh,
                                                   float* th_out) {
  __shared__ float xs[EMB];
  __shared__ float hs[HDIM];
  __shared__ float cs[HDIM];
  __shared__ float zs[4 * HDIM];
  int tid = threadIdx.x;
  if (tid < HDIM) { hs[tid] = 0.0f; cs[tid] = 0.0f; }
  __syncthreads();
  for (int t = 0; t < T_STEPS; ++t) {
    int tok = targets[t];
    for (int i = tid; i < EMB; i += 512) xs[i] = emb[(size_t)tok * EMB + i];
    __syncthreads();
    float z = bih[tid] + bhh[tid];
    const float* wr = wih + (size_t)tid * EMB;
    for (int k = 0; k < EMB; ++k) z += wr[k] * xs[k];
    const float* ur = whh + (size_t)tid * HDIM;
    for (int k = 0; k < HDIM; ++k) z += ur[k] * hs[k];
    zs[tid] = z;
    __syncthreads();
    if (tid < HDIM) {
      float iv = sigf(zs[tid]);
      float fv = sigf(zs[HDIM + tid]);
      float gv = tanhf(zs[2 * HDIM + tid]);
      float ov = sigf(zs[3 * HDIM + tid]);
      float c = fv * cs[tid] + iv * gv;
      cs[tid] = c;
      hs[tid] = ov * tanhf(c);
    }
    __syncthreads();
  }
  if (tid < HDIM) th_out[tid] = hs[tid];
}

// bias_tree[n] = bias + th contribution of the constant tt third of e2
__global__ void bias_tree_kernel(const float* Ui, const float* Ui_b,
                                 const float* Uo, const float* Uo_b,
                                 const float* Uu, const float* Uu_b,
                                 const float* Uf1_b, const float* Uf2_b,
                                 const float* th, float* bias_tree) {
  int j = threadIdx.x;
  if (j < 384) {
    int g = j >> 7, c = j & 127;
    const float* W = (g == 0) ? Ui : ((g == 1) ? Uo : Uu);
    const float* B = (g == 0) ? Ui_b : ((g == 1) ? Uo_b : Uu_b);
    float s = B[c];
    for (int k = 0; k < HDIM; ++k) s += th[k] * W[c * 384 + 256 + k];
    bias_tree[j] = s;
  } else if (j < 512) {
    bias_tree[j] = Uf1_b[j - 384];
  } else if (j < NTREE) {
    bias_tree[j] = Uf2_b[j - 512];
  }
}

// ---- leaf stage: 2 waves/block, M=32 per wave (B fragment reused 2x) -------

__global__ __launch_bounds__(64) void leaf_kernel(const half_t* embh, const int* tokens,
                                                  const half_t* Wpack, const float* bias_leaf,
                                                  half_t* hbuf, float* cbuf) {
  __shared__ half_t Ap[64 * AP_STRIDE];   // 64 rows x 328 halfs = 42 KB
  int tid = threadIdx.x;
  int rowBase = blockIdx.x * 64;
  __builtin_prefetch(Wpack, 0, 1);

  // stage 64 gathered f16 embedding rows (dword copies; K 300..319 zero pad)
  for (int idx = tid; idx < 64 * 160; idx += 64) {
    int r = idx / 160, k2 = idx - r * 160;          // k2 = dword index (2 halfs)
    unsigned v = 0u;
    if (k2 < 150) {
      int tok = tokens[rowBase + r];
      v = ((const unsigned*)embh)[(size_t)tok * 150 + k2];
    }
    ((unsigned*)(Ap + r * AP_STRIDE))[k2] = v;
  }
  __syncthreads();

  int wave = tid >> 5, lane = tid & 31;
  int m0 = wave * 32;                               // 2 M-tiles per wave
  int ln = lane & 15, hi = lane >> 4;

  v16h a[2][10];
#pragma unroll
  for (int kt = 0; kt < 10; ++kt) {
    a[0][kt] = load_afrag_lds(Ap, m0,      kt * 32, lane);
    a[1][kt] = load_afrag_lds(Ap, m0 + 16, kt * 32, lane);
  }

  for (int t = 0; t < 8; ++t) {
    int colg = t * 16 + ln;
    v8f zi[2], zo[2], zu[2];
    float b0 = bias_leaf[colg], b1 = bias_leaf[128 + colg], b2 = bias_leaf[256 + colg];
#pragma unroll
    for (int mt = 0; mt < 2; ++mt)
#pragma unroll
      for (int r = 0; r < 8; ++r) { zi[mt][r] = b0; zo[mt][r] = b1; zu[mt][r] = b2; }
#pragma unroll
    for (int kt = 0; kt < 10; ++kt) {
      v16h bI = load_bfrag_global(Wpack, NLEAF, kt * 32, t * 16, lane);
      zi[0] = wmma_f16(a[0][kt], bI, zi[0]);
      zi[1] = wmma_f16(a[1][kt], bI, zi[1]);
      v16h bO = load_bfrag_global(Wpack, NLEAF, kt * 32, 128 + t * 16, lane);
      zo[0] = wmma_f16(a[0][kt], bO, zo[0]);
      zo[1] = wmma_f16(a[1][kt], bO, zo[1]);
      v16h bU = load_bfrag_global(Wpack, NLEAF, kt * 32, 256 + t * 16, lane);
      zu[0] = wmma_f16(a[0][kt], bU, zu[0]);
      zu[1] = wmma_f16(a[1][kt], bU, zu[1]);
    }
#pragma unroll
    for (int mt = 0; mt < 2; ++mt) {
#pragma unroll
      for (int r = 0; r < 8; ++r) {
        int rowg = rowBase + m0 + mt * 16 + hi * 8 + r;
        float iv = sigf(zi[mt][r]);
        float ov = sigf(zo[mt][r]);
        float uv = fmaxf(zu[mt][r], 0.0f);
        float cv = iv * uv;
        float hv = ov * fmaxf(cv, 0.0f);
        hbuf[(size_t)rowg * HDIM + colg] = (half_t)hv;
        cbuf[(size_t)rowg * HDIM + colg] = cv;
      }
    }
  }
}

// ---- one tree level: 2 waves/block, M=32 per wave --------------------------

__global__ __launch_bounds__(64) void tree_kernel(const half_t* hprev, half_t* hcur,
                                                  const float* cprev, float* ccur,
                                                  const half_t* Upack, const float* bias_tree,
                                                  int n) {
  int tid = threadIdx.x;
  int wave = tid >> 5, lane = tid & 31;
  int m0 = blockIdx.x * 64 + wave * 32;             // 2 M-tiles per wave
  int ln = lane & 15, hi = lane >> 4;
  __builtin_prefetch(Upack, 0, 1);

  v16h a[2][8];
#pragma unroll
  for (int kt = 0; kt < 8; ++kt) {
    a[0][kt] = load_afrag_global(hprev, 2 * HDIM, m0,      kt * 32, lane, n);
    a[1][kt] = load_afrag_global(hprev, 2 * HDIM, m0 + 16, kt * 32, lane, n);
  }

  for (int t = 0; t < 8; ++t) {
    int col = t * 16 + ln;
    v8f acc[2][5];
#pragma unroll
    for (int g = 0; g < 5; ++g) {
      float b = bias_tree[(g * 8 + t) * 16 + ln];
#pragma unroll
      for (int mt = 0; mt < 2; ++mt)
#pragma unroll
        for (int r = 0; r < 8; ++r) acc[mt][g][r] = b;
    }
#pragma unroll
    for (int kt = 0; kt < 8; ++kt) {
#pragma unroll
      for (int g = 0; g < 5; ++g) {
        v16h b = load_bfrag_global(Upack, NTREE, kt * 32, (g * 8 + t) * 16, lane);
        acc[0][g] = wmma_f16(a[0][kt], b, acc[0][g]);
        acc[1][g] = wmma_f16(a[1][kt], b, acc[1][g]);
      }
    }
#pragma unroll
    for (int mt = 0; mt < 2; ++mt) {
#pragma unroll
      for (int r = 0; r < 8; ++r) {
        int row = m0 + mt * 16 + hi * 8 + r;
        if (row < n) {
          float lc = cprev[(size_t)(2 * row) * HDIM + col];
          float rc = cprev[(size_t)(2 * row + 1) * HDIM + col];
          float iv = sigf(acc[mt][0][r]);
          float ov = sigf(acc[mt][1][r]);
          float uv = fmaxf(acc[mt][2][r], 0.0f);
          float f1 = sigf(acc[mt][3][r]);
          float f2 = sigf(acc[mt][4][r]);
          float cv = iv * uv + f1 * lc + f2 * rc;
          float hv = ov * fmaxf(cv, 0.0f);
          hcur[(size_t)row * HDIM + col] = (half_t)hv;
          ccur[(size_t)row * HDIM + col] = cv;
        }
      }
    }
  }
}

// ---- final projection: out = all_h @ proj_w.T + proj_b ---------------------

__global__ void proj_kernel(const half_t* hbuf, const float* proj_w, const float* proj_b,
                            float* out, int total) {
  __shared__ float pw[3 * HDIM];
  __shared__ float pb[3];
  int tid = threadIdx.x;
  for (int i = tid; i < 3 * HDIM; i += blockDim.x) pw[i] = proj_w[i];
  if (tid < 3) pb[tid] = proj_b[tid];
  __syncthreads();
  int row = blockIdx.x * blockDim.x + tid;
  if (row >= total) return;
  const half_t* hp = hbuf + (size_t)row * HDIM;
  float s0 = pb[0], s1 = pb[1], s2 = pb[2];
  for (int k = 0; k < HDIM; ++k) {
    float hv = (float)hp[k];
    s0 += hv * pw[k];
    s1 += hv * pw[HDIM + k];
    s2 += hv * pw[2 * HDIM + k];
  }
  out[(size_t)row * 3 + 0] = s0;
  out[(size_t)row * 3 + 1] = s1;
  out[(size_t)row * 3 + 2] = s2;
}

// ---- host orchestration ----------------------------------------------------

extern "C" void kernel_launch(void* const* d_in, const int* in_sizes, int n_in,
                              void* d_out, int out_size, void* d_ws, size_t ws_size,
                              hipStream_t stream) {
  const float* emb      = (const float*)d_in[0];
  const float* Wi_w     = (const float*)d_in[1];
  const float* Wi_b     = (const float*)d_in[2];
  const float* Wo_w     = (const float*)d_in[3];
  const float* Wo_b     = (const float*)d_in[4];
  const float* Wu_w     = (const float*)d_in[5];
  const float* Wu_b     = (const float*)d_in[6];
  const float* Ui_w     = (const float*)d_in[7];
  const float* Ui_b     = (const float*)d_in[8];
  const float* Uo_w     = (const float*)d_in[9];
  const float* Uo_b     = (const float*)d_in[10];
  const float* Uu_w     = (const float*)d_in[11];
  const float* Uu_b     = (const float*)d_in[12];
  const float* Uf1_w    = (const float*)d_in[13];
  const float* Uf1_b    = (const float*)d_in[14];
  const float* Uf2_w    = (const float*)d_in[15];
  const float* Uf2_b    = (const float*)d_in[16];
  const float* lstm_wih = (const float*)d_in[17];
  const float* lstm_whh = (const float*)d_in[18];
  const float* lstm_bih = (const float*)d_in[19];
  const float* lstm_bhh = (const float*)d_in[20];
  const float* proj_w   = (const float*)d_in[21];
  const float* proj_b   = (const float*)d_in[22];
  const int*   tokens   = (const int*)d_in[23];
  const int*   targets  = (const int*)d_in[24];
  float* out = (float*)d_out;

  const int V_EMB = 50000 * EMB;   // 15,000,000

  // workspace carve-up (chunks after embh are only 128B-aligned)
  char* w = (char*)d_ws;
  half_t* hbuf = (half_t*)w;  w += (size_t)262144 * HDIM * sizeof(half_t);        // 64 MiB
  float*  cA   = (float*)w;   w += (size_t)L_LEAF * HDIM * sizeof(float);         // 64 MiB
  float*  cB   = (float*)w;   w += (size_t)(L_LEAF / 2) * HDIM * sizeof(float);   // 32 MiB
  half_t* embh = (half_t*)w;  w += (size_t)V_EMB * sizeof(half_t);                // 30 MB
  half_t* Wpack = (half_t*)w; w += (size_t)KLEAF * NLEAF * sizeof(half_t);
  half_t* Upack = (half_t*)w; w += (size_t)KTREE * NTREE * sizeof(half_t);
  float* bias_leaf = (float*)w; w += 384 * sizeof(float);
  float* bias_tree = (float*)w; w += NTREE * sizeof(float);
  float* th = (float*)w;        w += HDIM * sizeof(float);

  // embedding table -> f16 (once)
  emb_to_f16_kernel<<<(V_EMB / 2 + 255) / 256, 256, 0, stream>>>(emb, embh, V_EMB / 2);

  // weight packs (independent of LSTM)
  pack_leaf_kernel<<<(KLEAF * NLEAF + 255) / 256, 256, 0, stream>>>(
      Wi_w, Wo_w, Wu_w, Wi_b, Wo_b, Wu_b, Wpack, bias_leaf);
  pack_tree_kernel<<<(KTREE * NTREE + 255) / 256, 256, 0, stream>>>(
      Ui_w, Uo_w, Uu_w, Uf1_w, Uf2_w, Upack);

  // target LSTM -> th, then fold th into tree bias
  lstm_kernel<<<1, 512, 0, stream>>>(emb, targets, lstm_wih, lstm_whh, lstm_bih, lstm_bhh, th);
  bias_tree_kernel<<<1, NTREE, 0, stream>>>(Ui_w, Ui_b, Uo_w, Uo_b, Uu_w, Uu_b,
                                            Uf1_b, Uf2_b, th, bias_tree);

  // leaf stage
  leaf_kernel<<<L_LEAF / 64, 64, 0, stream>>>(embh, tokens, Wpack, bias_leaf, hbuf, cA);

  // tree reduction: 17 levels, h appended into hbuf, c ping-pongs cA/cB
  size_t prevOff = 0, curOff = L_LEAF;
  float* cP = cA;
  float* cC = cB;
  for (int nn = L_LEAF / 2; nn >= 1; nn >>= 1) {
    tree_kernel<<<(nn + 63) / 64, 64, 0, stream>>>(
        hbuf + prevOff * (size_t)HDIM, hbuf + curOff * (size_t)HDIM,
        cP, cC, Upack, bias_tree, nn);
    prevOff = curOff;
    curOff += nn;
    float* tmp = cP; cP = cC; cC = tmp;
  }

  // projection over all 262143 rows
  proj_kernel<<<(TOT_ROWS + 255) / 256, 256, 0, stream>>>(hbuf, proj_w, proj_b, out, TOT_ROWS);
}
